// UNetInnerProd_64278480552336
// MI455X (gfx1250) — matrix operand order repaired and checked by reading, not compiled, not run on hardware
//
#include <hip/hip_runtime.h>
#include <hip/hip_bf16.h>

typedef __attribute__((ext_vector_type(16))) _Float16 v16h;
typedef __attribute__((ext_vector_type(8)))  float    v8f;

#define FXc 48.0f
#define FYc 48.0f
#define CXc 48.0f
#define CYc 36.0f
#define WIMG 96
#define NPTS 6912
#define CCH 16
#define NTILES 432          /* NPTS / 16 */
#define JT 8                /* j-tiles per wave (A reuse) */
#define JGROUPS (NTILES/JT) /* 54 */

/* exp(-0.1*s) == exp2(EXPC*s), EXPC = -0.1*log2(e) */
#define EXPC (-0.14426950408889634f)

/* workspace layout (bytes, all 256-aligned) */
#define WS_ACCUM  0                          /* 2 floats: [0]=inner sum, [1]=norm sum */
#define WS_P1     256                        /* NPTS float4: xyz1, |xyz1|^2           */
#define WS_P2     (WS_P1 + NPTS*16)          /* NPTS float4: xyz2_trans               */
#define WS_P2N    (WS_P2 + NPTS*16)          /* NPTS float4: xyz2_trans_noisy         */
#define WS_APACK  (WS_P2N + NPTS*16)         /* NTILES*32 lanes * 16 halves           */
#define WS_BPACK  (WS_APACK + NTILES*32*16*2)

__global__ void init_kernel(float* accum) {
    if (threadIdx.x < 2) accum[threadIdx.x] = 0.0f;
}

__global__ void prep_kernel(const float* __restrict__ f1, const float* __restrict__ f2,
                            const float* __restrict__ dep1, const float* __restrict__ dep2,
                            const float* __restrict__ pose, const float* __restrict__ noise,
                            float4* __restrict__ p1, float4* __restrict__ p2,
                            float4* __restrict__ p2n,
                            _Float16* __restrict__ apack, _Float16* __restrict__ bpack,
                            float* __restrict__ accum) {
    int n = blockIdx.x * blockDim.x + threadIdx.x;
    if (n >= NPTS) return;

    // invK @ (u,v,1) for K=[[cx,fx,0],[cy,0,fy],[1,0,0]] is (1,(u-cx)/fx,(v-cy)/fy)
    int u = n % WIMG, v = n / WIMG;
    float ry = ((float)u - CXc) / FXc;
    float rz = ((float)v - CYc) / FYc;

    float d1 = dep1[n];
    float x1 = d1, y1 = d1 * ry, z1 = d1 * rz;
    p1[n] = make_float4(x1, y1, z1, x1*x1 + y1*y1 + z1*z1);

    float d2 = dep2[n];
    float hx = d2, hy = d2 * ry, hz = d2 * rz; // homo w = 1

    // pose_noisy = pose1_2 @ pose1_2_noise (rows 0..2 needed)
    float pn[12];
    #pragma unroll
    for (int r = 0; r < 3; ++r)
        #pragma unroll
        for (int c = 0; c < 4; ++c) {
            float s = 0.0f;
            #pragma unroll
            for (int k = 0; k < 4; ++k) s += pose[r*4 + k] * noise[k*4 + c];
            pn[r*4 + c] = s;
        }

    float tx = pose[0]*hx + pose[1]*hy + pose[2]*hz  + pose[3];
    float ty = pose[4]*hx + pose[5]*hy + pose[6]*hz  + pose[7];
    float tz = pose[8]*hx + pose[9]*hy + pose[10]*hz + pose[11];
    p2[n] = make_float4(tx, ty, tz, tx*tx + ty*ty + tz*tz);

    float nx = pn[0]*hx + pn[1]*hy + pn[2]*hz  + pn[3];
    float ny = pn[4]*hx + pn[5]*hy + pn[6]*hz  + pn[7];
    float nz = pn[8]*hx + pn[9]*hy + pn[10]*hz + pn[11];
    p2n[n] = make_float4(nx, ny, nz, nx*nx + ny*ny + nz*nz);

    // feature L2 norms + normalized f16 values
    float s1 = 0.0f, s2 = 0.0f;
    float c1[CCH], c2[CCH];
    #pragma unroll
    for (int c = 0; c < CCH; ++c) {
        c1[c] = f1[c*NPTS + n]; s1 += c1[c]*c1[c];
        c2[c] = f2[c*NPTS + n]; s2 += c2[c]*c2[c];
    }
    float n1 = sqrtf(s1), n2 = sqrtf(s2);
    float i1 = 1.0f / n1,  i2 = 1.0f / n2;
    atomicAdd(&accum[1], n1 + n2);

    int t = n >> 4, m = n & 15;
    // A layout (16-bit A 16x32): lanes 0-15 hold K=0..15 in halves 0..15,
    // lanes 16-31 hold K=16..31 -> zero padding.
    _Float16* aLo = apack + ((size_t)(t*32 + m     ) * 16);
    _Float16* aHi = apack + ((size_t)(t*32 + m + 16) * 16);
    #pragma unroll
    for (int h = 0; h < 8; ++h) {
        aLo[h]     = (_Float16)(c1[h]     * i1);
        aHi[h]     = (_Float16)0.0f;
        aLo[8 + h] = (_Float16)(c1[8 + h] * i1);
        aHi[8 + h] = (_Float16)0.0f;
    }
    // B layout (16-bit B 32x16): lanes 0-15 hold K=0..15 for column=lane,
    // lanes 16-31 hold K=16..31 -> zero padding.
    _Float16* bLo = bpack + ((size_t)(t*32 + m     ) * 16);
    _Float16* bHi = bpack + ((size_t)(t*32 + m + 16) * 16);
    #pragma unroll
    for (int k = 0; k < CCH; ++k) {
        bLo[k] = (_Float16)(c2[k] * i2);
        bHi[k] = (_Float16)0.0f;
    }
}

__global__ void __launch_bounds__(256)
fused_kernel(const float4* __restrict__ p1, const float4* __restrict__ p2,
             const float4* __restrict__ p2n,
             const v16h* __restrict__ apack, const v16h* __restrict__ bpack,
             float* __restrict__ accum) {
    int wave = (blockIdx.x * blockDim.x + threadIdx.x) >> 5;
    int lane = threadIdx.x & 31;
    int itile  = wave / JGROUPS;
    int jgroup = wave % JGROUPS;
    if (itile >= NTILES) return;

    v16h a = apack[itile * 32 + lane];

    int hi   = lane >> 4;
    int lcol = lane & 15;

    // rows for this lane's 8 accumulator elements are jj-invariant: hoist loads
    float4 q1r[8];
    #pragma unroll
    for (int p = 0; p < 8; ++p) q1r[p] = p1[itile * 16 + p + 8 * hi];

    float acc = 0.0f;

    for (int jj = 0; jj < JT; ++jj) {
        int jtile = jgroup * JT + jj;
        v16h b = bpack[jtile * 32 + lane];

        v8f c = {};
        c = __builtin_amdgcn_wmma_f32_16x16x32_f16(
                false, a, false, b, (short)0, c, false, false);

        int j = jtile * 16 + lcol;
        float4 q2  = p2[j];
        float4 q2n = p2n[j];

        #pragma unroll
        for (int p = 0; p < 8; ++p) {
            float4 q1 = q1r[p];
            float cr  = q1.x*q2.x  + q1.y*q2.y  + q1.z*q2.z;
            float d2  = fmaxf(q1.w + q2.w  - 2.0f*cr, 0.0f);
            float crn = q1.x*q2n.x + q1.y*q2n.y + q1.z*q2n.z;
            float d2n = fmaxf(q1.w + q2n.w - 2.0f*crn, 0.0f);
            // raw v_sqrt_f32 / v_exp_f32 (args in benign range; no fixups needed)
            float w   = __builtin_amdgcn_exp2f(EXPC * __builtin_amdgcn_sqrtf(d2));
            float wn  = __builtin_amdgcn_exp2f(EXPC * __builtin_amdgcn_sqrtf(d2n));
            acc += (w - wn) * c[p];
        }
    }

    // wave32 reduction
    #pragma unroll
    for (int s = 16; s > 0; s >>= 1) acc += __shfl_down(acc, s, 32);
    if (lane == 0) atomicAdd(&accum[0], acc);
}

__global__ void finalize_kernel(const float* __restrict__ accum, float* __restrict__ out) {
    float inner_neg = -accum[0];
    out[0] = inner_neg;           // final_loss
    out[1] = inner_neg;           // inner_neg
    out[2] = accum[1] * 100.0f;   // fea_norm_sum
}

extern "C" void kernel_launch(void* const* d_in, const int* in_sizes, int n_in,
                              void* d_out, int out_size, void* d_ws, size_t ws_size,
                              hipStream_t stream) {
    const float* f1    = (const float*)d_in[0];
    const float* f2    = (const float*)d_in[1];
    const float* dep1  = (const float*)d_in[2];
    const float* dep2  = (const float*)d_in[3];
    const float* pose  = (const float*)d_in[4];
    const float* noise = (const float*)d_in[5];
    float* out = (float*)d_out;

    char* ws = (char*)d_ws;
    float*    accum = (float*)   (ws + WS_ACCUM);
    float4*   p1    = (float4*)  (ws + WS_P1);
    float4*   p2    = (float4*)  (ws + WS_P2);
    float4*   p2n   = (float4*)  (ws + WS_P2N);
    _Float16* apack = (_Float16*)(ws + WS_APACK);
    _Float16* bpack = (_Float16*)(ws + WS_BPACK);

    init_kernel<<<1, 64, 0, stream>>>(accum);

    prep_kernel<<<(NPTS + 255) / 256, 256, 0, stream>>>(
        f1, f2, dep1, dep2, pose, noise, p1, p2, p2n, apack, bpack, accum);

    int waves  = NTILES * JGROUPS;           // 23328 waves
    int blocks = (waves * 32 + 255) / 256;   // 2916 blocks of 8 waves
    fused_kernel<<<blocks, 256, 0, stream>>>(
        p1, p2, p2n, (const v16h*)apack, (const v16h*)bpack, accum);

    finalize_kernel<<<1, 1, 0, stream>>>(accum, out);
}
